// LongTermMemory_76124000354476
// MI455X (gfx1250) — matrix operand order repaired and checked by reading, not compiled, not run on hardware
//
#include <hip/hip_runtime.h>
#include <math.h>
#include <stdint.h>

// ---------------------------------------------------------------------------
// LongTermMemory cross-attention for MI455X (gfx1250, wave32, WMMA).
//
// Pipeline (all bf16 matrix math, f32 accumulate, streaming softmax):
//   1) convert Wq/Wk/Wv/Wo -> bf16 in ws; init running (m,l,acc)
//   2) q = x @ Wq + bq                  (WMMA, store bf16)
//   3) for each of 16 chunks of 8192 memory rows:
//        k_c = mem_c @ Wk + bk         (WMMA, bf16)
//        v_c = mem_c @ Wv + bv         (WMMA, bf16)
//        S   = (q @ k_c^T) * 1/32      (WMMA, f32; A+B tiles async->LDS)
//        online softmax: m,l update, P = exp(S - m_new) bf16, acc *= alpha
//        acc += P @ v_c                (WMMA, f32 accumulate; A tile async)
//   4) retr = acc / l  (bf16)
//   5) o = retr @ Wo + bo + x          (WMMA, f32)
//   6) LayerNorm(o) -> d_out
//
// CDNA5 specifics: v_wmma_f32_16x16x32_bf16 with ds_load_b128 fragment feeds,
// global_load_async_to_lds_b128 (ASYNCcnt) for pure-copy tile staging,
// global_prefetch_b8 for the next K slice.
// ---------------------------------------------------------------------------

typedef __attribute__((ext_vector_type(16))) __bf16 bf16x16;
typedef __attribute__((ext_vector_type(8)))  __bf16 bf16x8;
typedef __attribute__((ext_vector_type(4)))  __bf16 bf16x4;
typedef __attribute__((ext_vector_type(8)))  float  f32x8;

#define D_DIM   1024
#define NQ      1024          // B*S = 2*512
#define MEMROWS 131072
#define CHUNK   8192
#define NCHUNK  16
#define TM      128
#define TN      128
#define TK      32
#define LDSPAD  8             // row stride 40 bf16 = 80 B (16-B multiple)

__device__ __forceinline__ __bf16 f2bf(float f) {
  union { float f; unsigned u; } in; in.f = f;
  unsigned u = in.u;
  u = (u + 0x7FFFu + ((u >> 16) & 1u)) >> 16;   // round-to-nearest-even
  union { unsigned short s; __bf16 b; } out; out.s = (unsigned short)u;
  return out.b;
}

// Async 16-byte global -> LDS copy (no VGPR landing; tracked by ASYNCcnt).
// LDS dest comes from the low 32 bits of the generic shared-memory address
// (ISA 10.2: LDS aperture addresses truncate to the LDS byte offset).
__device__ __forceinline__ void async_copy_b128(void* lds_ptr, const void* gptr) {
  unsigned lds_off = (unsigned)(uintptr_t)lds_ptr;
  unsigned long long ga = (unsigned long long)(uintptr_t)gptr;
  asm volatile("global_load_async_to_lds_b128 %0, %1, off"
               :: "v"(lds_off), "v"(ga) : "memory");
}
__device__ __forceinline__ void async_wait0() {
  asm volatile("s_wait_asynccnt 0x0" ::: "memory");
}

__device__ __forceinline__ float blockReduceMax(float v, float* sh) {
  __syncthreads();
  #pragma unroll
  for (int off = 16; off > 0; off >>= 1) v = fmaxf(v, __shfl_xor(v, off, 32));
  int lane = threadIdx.x & 31, wid = threadIdx.x >> 5;
  if (lane == 0) sh[wid] = v;
  __syncthreads();
  if (wid == 0) {
    float t = (lane < 8) ? sh[lane] : -__builtin_inff();
    #pragma unroll
    for (int off = 4; off > 0; off >>= 1) t = fmaxf(t, __shfl_xor(t, off, 32));
    if (lane == 0) sh[0] = t;
  }
  __syncthreads();
  return sh[0];
}

__device__ __forceinline__ float blockReduceSum(float v, float* sh) {
  __syncthreads();
  #pragma unroll
  for (int off = 16; off > 0; off >>= 1) v += __shfl_xor(v, off, 32);
  int lane = threadIdx.x & 31, wid = threadIdx.x >> 5;
  if (lane == 0) sh[wid] = v;
  __syncthreads();
  if (wid == 0) {
    float t = (lane < 8) ? sh[lane] : 0.f;
    #pragma unroll
    for (int off = 4; off > 0; off >>= 1) t += __shfl_xor(t, off, 32);
    if (lane == 0) sh[0] = t;
  }
  __syncthreads();
  return sh[0];
}

// ---------------------------------------------------------------------------
// Tiled WMMA GEMM: C[M,N] = A[M,K] @ B + epilogue.
//  AFMT:   0 = A is f32 (converted to bf16 on LDS stage-in, VGPR path)
//          1 = A is bf16 (pure copy -> global_load_async_to_lds_b128)
//  BTRANS: 0 = B stored [K][N] bf16 (weights/V) -> transpose on LDS store
//          1 = B stored [N][K] bf16 (key rows)  -> pure copy, async to LDS
//  EPI:    0 = store bf16, += bias[col]
//          1 = f32 accumulate into C (C += A@B)
//          2 = store f32, += bias[col] + res[row,col]   (out-proj + residual)
//          3 = store f32 scaled by `scale`              (attention scores)
// Workgroup: 256 threads = 8 waves (2x4), each wave owns a 64x32 sub-tile
// (4x2 fragments of 16x16), v_wmma_f32_16x16x32_bf16 with f32 accumulators.
// ---------------------------------------------------------------------------
template<int AFMT, int BTRANS, int EPI>
__global__ __launch_bounds__(256) void gemm_wmma_k(
    const void* __restrict__ Ap, const __bf16* __restrict__ Bp,
    const float* __restrict__ bias, const float* __restrict__ res,
    float scale, void* __restrict__ Cp,
    int M, int N, int K, int lda, int ldb, int ldc)
{
  (void)M; (void)N; (void)scale; (void)bias; (void)res;
  __shared__ __bf16 lds_a [TM][TK + LDSPAD];   // m-major A tile
  __shared__ __bf16 lds_bt[TN][TK + LDSPAD];   // n-major B tile (holds B^T)

  const int tid  = threadIdx.x;
  const int lane = tid & 31;
  const int wid  = tid >> 5;     // 0..7
  const int wm   = wid & 1;      // 64-row slab
  const int wn   = wid >> 1;     // 32-col slab
  const int lr   = lane & 15;
  const int lh   = lane >> 4;

  const int row0 = blockIdx.y * TM;
  const int col0 = blockIdx.x * TN;

  f32x8 acc[4][2];
  #pragma unroll
  for (int f = 0; f < 4; ++f)
    #pragma unroll
    for (int g = 0; g < 2; ++g)
      acc[f][g] = (f32x8){0.f,0.f,0.f,0.f,0.f,0.f,0.f,0.f};

  for (int k0 = 0; k0 < K; k0 += TK) {
    // ---- stage A tile (TM x TK) into LDS ----
    if (AFMT == 0) {
      // f32 source: load, convert to bf16 in VGPRs, ds_store
      const float* A = (const float*)Ap;
      int r = tid >> 3, cg = tid & 7;           // 8 x float4 per row
      #pragma unroll
      for (int rr = 0; rr < 4; ++rr) {
        float4 v = *reinterpret_cast<const float4*>(
            A + (size_t)(row0 + r) * lda + k0 + cg * 4);
        bf16x4 b; b[0]=f2bf(v.x); b[1]=f2bf(v.y); b[2]=f2bf(v.z); b[3]=f2bf(v.w);
        *reinterpret_cast<bf16x4*>(&lds_a[r][cg * 4]) = b;
        r += 32;
      }
    } else {
      // bf16 source: pure copy, async straight into LDS (4 x 16B per row)
      const __bf16* A = (const __bf16*)Ap;
      int r = tid >> 2, cg = tid & 3;
      #pragma unroll
      for (int rr = 0; rr < 2; ++rr) {
        async_copy_b128(&lds_a[r][cg * 8],
                        A + (size_t)(row0 + r) * lda + k0 + cg * 8);
        r += 64;
      }
    }
    // ---- stage B tile into n-major LDS ----
    if (BTRANS == 0) {
      // weights/V: B[K][N]; transpose on LDS store
      for (int i = tid; i < TK * TN; i += 256) {
        int kk = i >> 7, nn = i & 127;
        lds_bt[nn][kk] = Bp[(size_t)(k0 + kk) * ldb + col0 + nn];
      }
    } else {
      // key rows: B[N][K] already matches n-major layout -> async copy
      int n = tid >> 2, kg = tid & 3;
      #pragma unroll
      for (int rr = 0; rr < 2; ++rr) {
        async_copy_b128(&lds_bt[n][kg * 8],
                        Bp + (size_t)(col0 + n) * ldb + k0 + kg * 8);
        n += 64;
      }
    }
    // prefetch next A K-slice (emits global_prefetch_b8)
    if (lane == 0 && k0 + TK < K) {
      const char* Ab = (const char*)Ap;
      const size_t esz = (AFMT == 0) ? 4u : 2u;
      __builtin_prefetch(Ab + ((size_t)(row0 + wid * 16) * lda + k0 + TK) * esz, 0, 1);
    }
    if (AFMT == 1 || BTRANS == 1) async_wait0();  // ASYNCcnt not tracked by barrier
    __syncthreads();

    // ---- fragments (ISA 7.12.2 wave32 layouts), 16-B aligned LDS b128 loads
    bf16x16 afr[4];
    #pragma unroll
    for (int f = 0; f < 4; ++f) {
      int r = wm * 64 + f * 16 + lr;
      bf16x8 lo = *reinterpret_cast<const bf16x8*>(&lds_a[r][lh * 8]);      // K 0..7 / 8..15
      bf16x8 hi = *reinterpret_cast<const bf16x8*>(&lds_a[r][16 + lh * 8]); // K 16..23 / 24..31
      afr[f] = __builtin_shufflevector(lo, hi, 0,1,2,3,4,5,6,7,8,9,10,11,12,13,14,15);
    }
    bf16x16 bfr[2];
    #pragma unroll
    for (int g = 0; g < 2; ++g) {
      int c = wn * 32 + g * 16 + lr;
      bf16x8 lo = *reinterpret_cast<const bf16x8*>(&lds_bt[c][lh * 8]);      // K 0..7 / 8..15
      bf16x8 hi = *reinterpret_cast<const bf16x8*>(&lds_bt[c][16 + lh * 8]); // K 16..23 / 24..31
      bfr[g] = __builtin_shufflevector(lo, hi, 0,1,2,3,4,5,6,7,8,9,10,11,12,13,14,15);
    }

    #pragma unroll
    for (int f = 0; f < 4; ++f)
      #pragma unroll
      for (int g = 0; g < 2; ++g)
        acc[f][g] = __builtin_amdgcn_wmma_f32_16x16x32_bf16(
            false, afr[f], false, bfr[g], (short)0, acc[f][g], false, false);
    __syncthreads();
  }

  // ---- epilogue: C frag VGPR j -> row = base + lh*8 + j, col = base + lr
  #pragma unroll
  for (int f = 0; f < 4; ++f) {
    #pragma unroll
    for (int g = 0; g < 2; ++g) {
      #pragma unroll
      for (int j = 0; j < 8; ++j) {
        int crow = row0 + wm * 64 + f * 16 + lh * 8 + j;
        int ccol = col0 + wn * 32 + g * 16 + lr;
        size_t idx = (size_t)crow * ldc + ccol;
        float v = acc[f][g][j];
        if (EPI == 0) {
          ((__bf16*)Cp)[idx] = f2bf(v + bias[ccol]);
        } else if (EPI == 1) {
          ((float*)Cp)[idx] += v;
        } else if (EPI == 2) {
          ((float*)Cp)[idx] = v + bias[ccol] + res[idx];
        } else {
          ((float*)Cp)[idx] = v * scale;
        }
      }
    }
  }
}

// --- online softmax chunk update: max/sum reduce, P = exp(S-m_new), acc *= a
__global__ __launch_bounds__(256) void softmax_update_k(
    const float* __restrict__ S, __bf16* __restrict__ P,
    float* __restrict__ m_run, float* __restrict__ l_run,
    float* __restrict__ acc, int chunkN)
{
  __shared__ float sh[8];
  const int row = blockIdx.x, tid = threadIdx.x;
  const float* Sr = S + (size_t)row * chunkN;
  __bf16* Pr = P + (size_t)row * chunkN;

  float mx = -__builtin_inff();
  for (int i = tid; i < chunkN; i += 256) mx = fmaxf(mx, Sr[i]);
  mx = blockReduceMax(mx, sh);

  const float m_old = m_run[row];
  const float l_old = l_run[row];
  const float m_new = fmaxf(m_old, mx);

  float s = 0.f;
  for (int i = tid; i < chunkN; i += 256) {
    float e = __expf(Sr[i] - m_new);
    s += e;
    Pr[i] = f2bf(e);
  }
  s = blockReduceSum(s, sh);

  const float alpha = __expf(m_old - m_new);   // 0 on first chunk (m_old=-inf)
  float* ar = acc + (size_t)row * D_DIM;
  for (int d = tid; d < D_DIM; d += 256) ar[d] *= alpha;
  if (tid == 0) { m_run[row] = m_new; l_run[row] = l_old * alpha + s; }
}

__global__ __launch_bounds__(256) void init_k(float* acc, float* m_run, float* l_run) {
  int idx = blockIdx.x * 256 + threadIdx.x;
  acc[idx] = 0.f;
  if (idx < NQ) { m_run[idx] = -__builtin_inff(); l_run[idx] = 0.f; }
}

__global__ __launch_bounds__(256) void cvt_k(const float* __restrict__ src,
                                             __bf16* __restrict__ dst, int n) {
  int idx = blockIdx.x * 256 + threadIdx.x;
  if (idx < n) dst[idx] = f2bf(src[idx]);
}

__global__ __launch_bounds__(256) void finalize_k(const float* __restrict__ acc,
                                                  const float* __restrict__ l_run,
                                                  __bf16* __restrict__ retr) {
  int idx = blockIdx.x * 256 + threadIdx.x;
  int row = idx >> 10;                      // D_DIM == 1024
  retr[idx] = f2bf(acc[idx] / l_run[row]);
}

__global__ __launch_bounds__(256) void layernorm_k(
    const float* __restrict__ o, const float* __restrict__ gamma,
    const float* __restrict__ beta, float* __restrict__ out)
{
  __shared__ float sh[8];
  const int row = blockIdx.x, tid = threadIdx.x;
  const float* orow = o + (size_t)row * D_DIM;
  float s = 0.f, s2 = 0.f;
  for (int d = tid; d < D_DIM; d += 256) { float v = orow[d]; s += v; s2 += v * v; }
  s  = blockReduceSum(s,  sh);
  s2 = blockReduceSum(s2, sh);
  const float mu   = s  * (1.f / D_DIM);
  const float var  = s2 * (1.f / D_DIM) - mu * mu;
  const float rstd = rsqrtf(var + 1e-5f);
  float* orow_out = out + (size_t)row * D_DIM;
  for (int d = tid; d < D_DIM; d += 256)
    orow_out[d] = (orow[d] - mu) * rstd * gamma[d] + beta[d];
}

extern "C" void kernel_launch(void* const* d_in, const int* in_sizes, int n_in,
                              void* d_out, int out_size, void* d_ws, size_t ws_size,
                              hipStream_t stream) {
  (void)in_sizes; (void)n_in; (void)out_size; (void)ws_size;
  const float* x    = (const float*)d_in[0];
  const float* memb = (const float*)d_in[1];
  const float* Wq   = (const float*)d_in[2];  const float* bq = (const float*)d_in[3];
  const float* Wk   = (const float*)d_in[4];  const float* bk = (const float*)d_in[5];
  const float* Wv   = (const float*)d_in[6];  const float* bv = (const float*)d_in[7];
  const float* Wo   = (const float*)d_in[8];  const float* bo = (const float*)d_in[9];
  const float* g    = (const float*)d_in[10]; const float* b  = (const float*)d_in[11];

  // workspace layout (~101 MB)
  const size_t MB = 1024ull * 1024ull;
  char* w = (char*)d_ws;
  __bf16* Wq_b  = (__bf16*)(w + 0 * MB);
  __bf16* Wk_b  = (__bf16*)(w + 2 * MB);
  __bf16* Wv_b  = (__bf16*)(w + 4 * MB);
  __bf16* Wo_b  = (__bf16*)(w + 6 * MB);
  __bf16* q_b   = (__bf16*)(w + 8 * MB);
  __bf16* k_b   = (__bf16*)(w + 10 * MB);   // 16 MB (CHUNK x D bf16)
  __bf16* v_b   = (__bf16*)(w + 26 * MB);   // 16 MB
  float*  Sbuf  = (float*) (w + 42 * MB);   // 32 MB (NQ x CHUNK f32)
  __bf16* Pbuf  = (__bf16*)(w + 74 * MB);   // 16 MB
  float*  accb  = (float*) (w + 90 * MB);   //  4 MB (NQ x D f32)
  float*  m_run = (float*) (w + 94 * MB);
  float*  l_run = (float*) (w + 94 * MB + 8192);
  __bf16* retr  = (__bf16*)(w + 95 * MB);   //  2 MB
  float*  opre  = (float*) (w + 97 * MB);   //  4 MB

  const dim3 blk(256);
  const int DD = D_DIM * D_DIM;             // 1,048,576

  cvt_k<<<4096, blk, 0, stream>>>(Wq, Wq_b, DD);
  cvt_k<<<4096, blk, 0, stream>>>(Wk, Wk_b, DD);
  cvt_k<<<4096, blk, 0, stream>>>(Wv, Wv_b, DD);
  cvt_k<<<4096, blk, 0, stream>>>(Wo, Wo_b, DD);
  init_k<<<4096, blk, 0, stream>>>(accb, m_run, l_run);

  // q = x @ Wq + bq   -> bf16
  gemm_wmma_k<0, 0, 0><<<dim3(D_DIM / TN, NQ / TM), blk, 0, stream>>>(
      x, Wq_b, bq, nullptr, 1.f, q_b, NQ, D_DIM, D_DIM, D_DIM, D_DIM, D_DIM);

  const float scl = 0.03125f;               // 1/sqrt(1024)
  for (int c = 0; c < NCHUNK; ++c) {
    const float* memc = memb + (size_t)c * CHUNK * D_DIM;
    gemm_wmma_k<0, 0, 0><<<dim3(D_DIM / TN, CHUNK / TM), blk, 0, stream>>>(
        memc, Wk_b, bk, nullptr, 1.f, k_b, CHUNK, D_DIM, D_DIM, D_DIM, D_DIM, D_DIM);
    gemm_wmma_k<0, 0, 0><<<dim3(D_DIM / TN, CHUNK / TM), blk, 0, stream>>>(
        memc, Wv_b, bv, nullptr, 1.f, v_b, CHUNK, D_DIM, D_DIM, D_DIM, D_DIM, D_DIM);
    // S = scale * q @ k^T   (both tiles async->LDS)
    gemm_wmma_k<1, 1, 3><<<dim3(CHUNK / TN, NQ / TM), blk, 0, stream>>>(
        q_b, k_b, nullptr, nullptr, scl, Sbuf, NQ, CHUNK, D_DIM, D_DIM, D_DIM, CHUNK);
    softmax_update_k<<<NQ, blk, 0, stream>>>(Sbuf, Pbuf, m_run, l_run, accb, CHUNK);
    // acc += P @ v
    gemm_wmma_k<1, 0, 1><<<dim3(D_DIM / TN, NQ / TM), blk, 0, stream>>>(
        Pbuf, v_b, nullptr, nullptr, 1.f, accb, NQ, D_DIM, CHUNK, CHUNK, D_DIM, D_DIM);
  }

  finalize_k<<<4096, blk, 0, stream>>>(accb, l_run, retr);
  // o = retr @ Wo + bo + x
  gemm_wmma_k<1, 0, 2><<<dim3(D_DIM / TN, NQ / TM), blk, 0, stream>>>(
      retr, Wo_b, bo, x, 1.f, opre, NQ, D_DIM, D_DIM, D_DIM, D_DIM, D_DIM);
  layernorm_k<<<NQ, blk, 0, stream>>>(opre, g, b, (float*)d_out);
}
